// LearnedGateWaveLayer_73641509257842
// MI455X (gfx1250) — matrix operand order repaired and checked by reading, not compiled, not run on hardware
//
#include <hip/hip_runtime.h>

// ============================================================================
// LearnedGateWaveLayer on MI455X (gfx1250)
//
//   out[b] = ( irfft( gate ⊙ rfft(x[b], axis=seq) ) ) @ W^T + bias
//
// Spectral pipeline along seq (N=2048) is a fixed linear map => dense GEMMs
// on the WMMA units (v_wmma_f32_16x16x32_f16, f32 accumulate):
//   GEMM1: Xr = Cf·x[b], Xi = Sf·x[b]; epilogue gate -> Yr, Yi (f16)
//   GEMM2: y  = Ci·Yr + Si·Yi  (inverse rDFT)
//   GEMM3: out[b] = y·W^T + bias
// ~450 GFLOP in WMMA; intermediates fit in L2 (192 MB) so HBM traffic is
// ~270 MB (~12us @ 23.3 TB/s) => matrix-core bound.
//
// CDNA5 data movement:
//  - GEMM2 A tiles (Ci/Si, 128x32 f16) are moved by the Tensor Data Mover:
//    one tensor_load_to_lds per tile (D# descriptor, TENSORcnt), with TDM
//    padding reproducing the 40-half LDS row stride.
//  - GEMM1/GEMM3 A tiles use GLOBAL_LOAD_ASYNC_TO_LDS_B128 (ASYNCcnt).
//  - All staging is double-buffered in LDS and overlaps the WMMA chain;
//    fenced with s_wait_asynccnt / s_wait_tensorcnt before the barrier.
//  - B tiles need f32->f16 cvt or LDS-side transpose => register path.
// ============================================================================

constexpr int SEQ = 2048;
constexpr int HID = 2048;
constexpr int NF  = 1025;   // SEQ/2 + 1
constexpr int FP  = 1152;   // freq dim padded to 9*128

typedef _Float16 v16h __attribute__((ext_vector_type(16)));
typedef _Float16 v8h  __attribute__((ext_vector_type(8)));
typedef float    v8f  __attribute__((ext_vector_type(8)));
typedef unsigned int u32x4 __attribute__((ext_vector_type(4)));
typedef int          i32x4 __attribute__((ext_vector_type(4)));
typedef int          i32x8 __attribute__((ext_vector_type(8)));

#define LDA 40  // LDS row stride (halves): 80B = 20 banks -> conflict-free frags

#define WMMA(A, B, C) __builtin_amdgcn_wmma_f32_16x16x32_f16( \
    false, (A), false, (B), (short)0, (C), false, false)

union V16 { v16h v; v8h h[2]; };

// --- raw 32-bit LDS offset of a (generic) pointer into __shared__ ----------
__device__ __forceinline__ unsigned lds_off(const void* p) {
  return (unsigned)(unsigned long long)(__attribute__((address_space(3))) const void*)p;
}

// --- async memory->LDS copy, 16 bytes per lane (GV mode, SADDR=off) --------
__device__ __forceinline__ void async_b128(const _Float16* g, _Float16* l) {
  asm volatile("global_load_async_to_lds_b128 %0, %1, off"
               :: "v"(lds_off(l)), "v"(g) : "memory");
}
__device__ __forceinline__ void wait_async0() {
  asm volatile("s_wait_asynccnt 0x0" ::: "memory");
}

// --- TDM: move a tileRowsx32-half tile of a row-major f16 matrix into LDS --
// D# per ISA §8.3/8.4. LDS rows get 16B pad after each 64B row => stride LDA.
// Descriptor is wave-uniform; issue from one wave only.
__device__ __forceinline__ void tdm_load_tile_f16(const _Float16* gtile,
                                                  unsigned ldsAddr,
                                                  unsigned ld /*elems*/,
                                                  unsigned tileRows) {
  const unsigned long long ga = (unsigned long long)gtile;
  u32x4 g0;
  g0[0] = 1u;                                  // count=1 (valid), user mode
  g0[1] = ldsAddr;                             // lds_addr [63:32]
  g0[2] = (unsigned)ga;                        // global_addr [95:64]
  g0[3] = (unsigned)((ga >> 32) & 0x01FFFFFFu) // global_addr [120:96]
        | (2u << 30);                          // type=2 ("image") [127:126]
  i32x8 g1;
  g1[0] = (int)(0x00010000u       // data_size=1 -> 2 bytes
              | 0x00100000u       // pad_enable
              | (3u << 22)        // pad_interval: 16 DWORDs (64B) between pads
              | (3u << 25));      // pad_amount:   4 DWORDs (16B)
  g1[1] = (int)((ld & 0xFFFFu) << 16);         // tensor_dim0[15:0]
  g1[2] = (int)((ld >> 16) | (0u << 16));      // tensor_dim0[31:16], dim1[15:0]
  g1[3] = (int)(16u | (32u << 16));            // tensor_dim1=1<<20 (hi), tile_dim0=32
  g1[4] = (int)(tileRows & 0xFFFFu);           // tile_dim1
  g1[5] = (int)ld;                             // tensor_dim0_stride[31:0]
  g1[6] = 0;                                   // stride hi, dim1_stride lo
  g1[7] = 0;
  const i32x4 z4 = {0, 0, 0, 0};
#if __clang_major__ >= 23
  const i32x8 z8 = {0, 0, 0, 0, 0, 0, 0, 0};
  __builtin_amdgcn_tensor_load_to_lds(g0, g1, z4, z4, z8, 0);
#else
  __builtin_amdgcn_tensor_load_to_lds(g0, g1, z4, z4, 0);
#endif
}

// A fragment, 16x32 f16 tile (ISA 7.12.2): lane<16 -> M=lane, K in {0..7,16..23};
// lane>=16 -> M=lane-16, K in {8..15,24..31}. LDS row-major [m][k], stride LDA.
__device__ __forceinline__ v16h frag_a(const _Float16* sA, int mBase, int lane) {
  const int m  = mBase + (lane & 15);
  const int kb = (lane & 16) ? 8 : 0;
  const _Float16* p = sA + m * LDA + kb;
  V16 u;
  u.h[0] = *(const v8h*)(p);
  u.h[1] = *(const v8h*)(p + 16);
  return u.v;
}

// B fragment, 32x16 f16 tile: lane<16 -> N=lane, K in {0..7,16..23}; lane>=16 -> N=lane-16,
// K in {8..15,24..31}. LDS holds B transposed [n][k], stride LDA.
__device__ __forceinline__ v16h frag_b(const _Float16* sB, int nBase, int lane) {
  const int n  = nBase + (lane & 15);
  const int kb = (lane & 16) ? 8 : 0;
  const _Float16* p = sB + n * LDA + kb;
  V16 u;
  u.h[0] = *(const v8h*)(p);
  u.h[1] = *(const v8h*)(p + 16);
  return u.v;
}

// ---------------------------------------------------------------------------
// Build f16 DFT matrices. Twiddle index reduced mod 2048 in integers => exact.
// ---------------------------------------------------------------------------
__global__ __launch_bounds__(256)
void dft_setup_kernel(_Float16* __restrict__ Cf, _Float16* __restrict__ Sf,
                      _Float16* __restrict__ Ci, _Float16* __restrict__ Si) {
  const int idx = blockIdx.x * 256 + threadIdx.x;  // covers FP*SEQ exactly
  const float TW = 0.0030679615757712823f;         // 2*pi / 2048
  {
    const int f = idx >> 11, s = idx & 2047;
    float c = 0.f, sn = 0.f;
    if (f < NF) {
      const int tt = (f * s) & 2047;
      __sincosf((float)tt * TW, &sn, &c);
    }
    Cf[idx] = (_Float16)c;
    Sf[idx] = (_Float16)(-sn);
  }
  {
    const int s = idx / FP, f = idx - s * FP;
    float cv = 0.f, sv = 0.f;
    if (f < NF) {
      const float w  = (f == 0 || f == NF - 1) ? 1.0f : 2.0f;
      const float sc = w * (1.0f / 2048.0f);
      const int tt = (f * s) & 2047;
      float sn, c;
      __sincosf((float)tt * TW, &sn, &c);
      cv = c * sc;
      sv = -sn * sc;
    }
    Ci[idx] = (_Float16)cv;
    Si[idx] = (_Float16)sv;
  }
}

// ---------------------------------------------------------------------------
// GEMM1: Xr = Cf @ x[b], Xi = Sf @ x[b]; fused gate -> Yr, Yi (f16).
// Block tile 128x64, 8 waves (4x2), 32x32 per wave, K-step 32, double-buffered.
// A tiles via async-to-LDS.
// ---------------------------------------------------------------------------
__global__ __launch_bounds__(256)
void fwd_gate_gemm(const _Float16* __restrict__ Cf, const _Float16* __restrict__ Sf,
                   const float* __restrict__ xb,
                   const float* __restrict__ gr, const float* __restrict__ gi,
                   _Float16* __restrict__ Yr, _Float16* __restrict__ Yi) {
  constexpr int OAI = 128 * LDA, OB = 2 * 128 * LDA, BUF = 2 * 128 * LDA + 64 * LDA;
  __shared__ _Float16 smem[2 * BUF];

  const int t = threadIdx.x, lane = t & 31, wid = t >> 5;
  const int wm = wid >> 1, wn = wid & 1;
  const int m0 = blockIdx.y * 128, n0 = blockIdx.x * 64;

  const int ar = t >> 1, ac = (t & 1) * 16;   // A staging: 2 threads/row
  const int bk = t >> 3, bc = (t & 7) * 8;    // B staging: 8 threads/k-row

  v8f accR[2][2] = {}, accI[2][2] = {};

  auto stage = [&](int base, int kk) {
    _Float16* sAr = smem + base;
    _Float16* sAi = smem + base + OAI;
    _Float16* sB  = smem + base + OB;
    const _Float16* gC = Cf + (size_t)(m0 + ar) * SEQ + kk + ac;
    const _Float16* gS = Sf + (size_t)(m0 + ar) * SEQ + kk + ac;
    async_b128(gC,     sAr + ar * LDA + ac);
    async_b128(gC + 8, sAr + ar * LDA + ac + 8);
    async_b128(gS,     sAi + ar * LDA + ac);
    async_b128(gS + 8, sAi + ar * LDA + ac + 8);
    // B tile: f32 -> f16 with transpose: sB[n][k] = x[kk+k][n0+n]
    const float* xp = xb + (size_t)(kk + bk) * HID + n0 + bc;
    const float4 x0 = *(const float4*)(xp);
    const float4 x1 = *(const float4*)(xp + 4);
    _Float16* d = sB + bk;
    d[(bc + 0) * LDA] = (_Float16)x0.x;
    d[(bc + 1) * LDA] = (_Float16)x0.y;
    d[(bc + 2) * LDA] = (_Float16)x0.z;
    d[(bc + 3) * LDA] = (_Float16)x0.w;
    d[(bc + 4) * LDA] = (_Float16)x1.x;
    d[(bc + 5) * LDA] = (_Float16)x1.y;
    d[(bc + 6) * LDA] = (_Float16)x1.z;
    d[(bc + 7) * LDA] = (_Float16)x1.w;
  };

  stage(0, 0);
  wait_async0();
  __syncthreads();

  int buf = 0;
  for (int kk = 0; kk < SEQ; kk += 32) {
    if (kk + 32 < SEQ) stage((buf ^ 1) * BUF, kk + 32);  // overlap with compute

    const _Float16* sAr = smem + buf * BUF;
    const _Float16* sAi = smem + buf * BUF + OAI;
    const _Float16* sB  = smem + buf * BUF + OB;

    const v16h a_r0 = frag_a(sAr, wm * 32,      lane);
    const v16h a_r1 = frag_a(sAr, wm * 32 + 16, lane);
    const v16h a_i0 = frag_a(sAi, wm * 32,      lane);
    const v16h a_i1 = frag_a(sAi, wm * 32 + 16, lane);
    const v16h b0   = frag_b(sB,  wn * 32,      lane);
    const v16h b1   = frag_b(sB,  wn * 32 + 16, lane);

    accR[0][0] = WMMA(a_r0, b0, accR[0][0]);
    accR[0][1] = WMMA(a_r0, b1, accR[0][1]);
    accR[1][0] = WMMA(a_r1, b0, accR[1][0]);
    accR[1][1] = WMMA(a_r1, b1, accR[1][1]);
    accI[0][0] = WMMA(a_i0, b0, accI[0][0]);
    accI[0][1] = WMMA(a_i0, b1, accI[0][1]);
    accI[1][0] = WMMA(a_i1, b0, accI[1][0]);
    accI[1][1] = WMMA(a_i1, b1, accI[1][1]);

    wait_async0();
    __syncthreads();
    buf ^= 1;
  }

  // Gate epilogue: (Xr + iXi)(gr + igi) -> f16 Yr/Yi; padded rows (f>=NF) -> 0.
  #pragma unroll
  for (int mt = 0; mt < 2; ++mt)
    #pragma unroll
    for (int nt = 0; nt < 2; ++nt)
      #pragma unroll
      for (int r = 0; r < 8; ++r) {
        const int f = m0 + wm * 32 + mt * 16 + ((lane & 16) ? r + 8 : r);
        const int h = n0 + wn * 32 + nt * 16 + (lane & 15);
        float g_r = 0.f, g_i = 0.f;
        if (f < NF) {
          g_r = gr[(size_t)f * HID + h];
          g_i = gi[(size_t)f * HID + h];
        }
        const float xr = accR[mt][nt][r], xi = accI[mt][nt][r];
        Yr[(size_t)f * HID + h] = (_Float16)(xr * g_r - xi * g_i);
        Yi[(size_t)f * HID + h] = (_Float16)(xr * g_i + xi * g_r);
      }
}

// ---------------------------------------------------------------------------
// GEMM2: y = Ci @ Yr + Si @ Yi. A tiles (Ci/Si) are moved by the TDM:
// one tensor_load_to_lds per 128x32 tile, issued by wave 0, TENSORcnt fenced.
// ---------------------------------------------------------------------------
__global__ __launch_bounds__(256)
void inv_gemm(const _Float16* __restrict__ Ci, const _Float16* __restrict__ Si,
              const _Float16* __restrict__ Yr, const _Float16* __restrict__ Yi,
              _Float16* __restrict__ y16) {
  constexpr int OAI = 128 * LDA, OBR = 2 * 128 * LDA, OBI = 2 * 128 * LDA + 64 * LDA;
  constexpr int BUF = 2 * 128 * LDA + 2 * 64 * LDA;
  __shared__ _Float16 smem[2 * BUF];

  const int t = threadIdx.x, lane = t & 31, wid = t >> 5;
  const int wm = wid >> 1, wn = wid & 1;
  const int m0 = blockIdx.y * 128, n0 = blockIdx.x * 64;

  const int bk = t >> 3, bc = (t & 7) * 8;

  v8f acc[2][2] = {};

  auto stage = [&](int base, int kk) {
    _Float16* sAr = smem + base;
    _Float16* sAi = smem + base + OAI;
    _Float16* sBr = smem + base + OBR;
    _Float16* sBi = smem + base + OBI;
    // A tiles: Tensor Data Mover (wave 0 issues; wave-uniform descriptors)
    if (wid == 0) {
      tdm_load_tile_f16(Ci + (size_t)m0 * FP + kk, lds_off(sAr), FP, 128);
      tdm_load_tile_f16(Si + (size_t)m0 * FP + kk, lds_off(sAi), FP, 128);
    }
    // B tiles: f16 with LDS-side transpose
    const v8h yr8 = *(const v8h*)(Yr + (size_t)(kk + bk) * HID + n0 + bc);
    const v8h yi8 = *(const v8h*)(Yi + (size_t)(kk + bk) * HID + n0 + bc);
    #pragma unroll
    for (int j = 0; j < 8; ++j) {
      sBr[(bc + j) * LDA + bk] = yr8[j];
      sBi[(bc + j) * LDA + bk] = yi8[j];
    }
  };

  stage(0, 0);
  __builtin_amdgcn_s_wait_tensorcnt(0);
  __syncthreads();

  int buf = 0;
  for (int kk = 0; kk < FP; kk += 32) {
    if (kk + 32 < FP) stage((buf ^ 1) * BUF, kk + 32);

    const _Float16* sAr = smem + buf * BUF;
    const _Float16* sAi = smem + buf * BUF + OAI;
    const _Float16* sBr = smem + buf * BUF + OBR;
    const _Float16* sBi = smem + buf * BUF + OBI;

    const v16h a_r0 = frag_a(sAr, wm * 32,      lane);
    const v16h a_r1 = frag_a(sAr, wm * 32 + 16, lane);
    const v16h a_i0 = frag_a(sAi, wm * 32,      lane);
    const v16h a_i1 = frag_a(sAi, wm * 32 + 16, lane);
    const v16h b_r0 = frag_b(sBr, wn * 32,      lane);
    const v16h b_r1 = frag_b(sBr, wn * 32 + 16, lane);
    const v16h b_i0 = frag_b(sBi, wn * 32,      lane);
    const v16h b_i1 = frag_b(sBi, wn * 32 + 16, lane);

    acc[0][0] = WMMA(a_r0, b_r0, acc[0][0]);
    acc[0][0] = WMMA(a_i0, b_i0, acc[0][0]);
    acc[0][1] = WMMA(a_r0, b_r1, acc[0][1]);
    acc[0][1] = WMMA(a_i0, b_i1, acc[0][1]);
    acc[1][0] = WMMA(a_r1, b_r0, acc[1][0]);
    acc[1][0] = WMMA(a_i1, b_i0, acc[1][0]);
    acc[1][1] = WMMA(a_r1, b_r1, acc[1][1]);
    acc[1][1] = WMMA(a_i1, b_i1, acc[1][1]);

    __builtin_amdgcn_s_wait_tensorcnt(0);
    __syncthreads();
    buf ^= 1;
  }

  #pragma unroll
  for (int mt = 0; mt < 2; ++mt)
    #pragma unroll
    for (int nt = 0; nt < 2; ++nt)
      #pragma unroll
      for (int r = 0; r < 8; ++r) {
        const int s = m0 + wm * 32 + mt * 16 + ((lane & 16) ? r + 8 : r);
        const int h = n0 + wn * 32 + nt * 16 + (lane & 15);
        y16[(size_t)s * HID + h] = (_Float16)acc[mt][nt][r];
      }
}

// ---------------------------------------------------------------------------
// GEMM3: out[b] = y @ W^T + bias.  sB[o][h] = W[o][h] (row copy + cvt).
// A tiles via async-to-LDS.
// ---------------------------------------------------------------------------
__global__ __launch_bounds__(256)
void proj_gemm(const _Float16* __restrict__ y16, const float* __restrict__ W,
               const float* __restrict__ bias, float* __restrict__ outb) {
  constexpr int OB = 128 * LDA, BUF = 128 * LDA + 64 * LDA;  // 7680 halves/buf
  __shared__ _Float16 smem[2 * BUF];

  const int t = threadIdx.x, lane = t & 31, wid = t >> 5;
  const int wm = wid >> 1, wn = wid & 1;
  const int m0 = blockIdx.y * 128, n0 = blockIdx.x * 64;

  const int ar = t >> 1, ac = (t & 1) * 16;
  const int bo = t >> 2, bh = (t & 3) * 8;    // B staging: 4 threads/o-row

  v8f acc[2][2] = {};

  auto stage = [&](int base, int kk) {
    _Float16* sA = smem + base;
    _Float16* sB = smem + base + OB;
    const _Float16* gA = y16 + (size_t)(m0 + ar) * HID + kk + ac;
    async_b128(gA,     sA + ar * LDA + ac);
    async_b128(gA + 8, sA + ar * LDA + ac + 8);
    const float* wp = W + (size_t)(n0 + bo) * HID + kk + bh;
    const float4 w0 = *(const float4*)(wp);
    const float4 w1 = *(const float4*)(wp + 4);
    v8h wv;
    wv[0] = (_Float16)w0.x; wv[1] = (_Float16)w0.y;
    wv[2] = (_Float16)w0.z; wv[3] = (_Float16)w0.w;
    wv[4] = (_Float16)w1.x; wv[5] = (_Float16)w1.y;
    wv[6] = (_Float16)w1.z; wv[7] = (_Float16)w1.w;
    *(v8h*)(sB + bo * LDA + bh) = wv;
  };

  stage(0, 0);
  wait_async0();
  __syncthreads();

  int buf = 0;
  for (int kk = 0; kk < HID; kk += 32) {
    if (kk + 32 < HID) stage((buf ^ 1) * BUF, kk + 32);

    const _Float16* sA = smem + buf * BUF;
    const _Float16* sB = smem + buf * BUF + OB;

    const v16h a0 = frag_a(sA, wm * 32,      lane);
    const v16h a1 = frag_a(sA, wm * 32 + 16, lane);
    const v16h b0 = frag_b(sB, wn * 32,      lane);
    const v16h b1 = frag_b(sB, wn * 32 + 16, lane);

    acc[0][0] = WMMA(a0, b0, acc[0][0]);
    acc[0][1] = WMMA(a0, b1, acc[0][1]);
    acc[1][0] = WMMA(a1, b0, acc[1][0]);
    acc[1][1] = WMMA(a1, b1, acc[1][1]);

    wait_async0();
    __syncthreads();
    buf ^= 1;
  }

  #pragma unroll
  for (int mt = 0; mt < 2; ++mt)
    #pragma unroll
    for (int nt = 0; nt < 2; ++nt)
      #pragma unroll
      for (int r = 0; r < 8; ++r) {
        const int s = m0 + wm * 32 + mt * 16 + ((lane & 16) ? r + 8 : r);
        const int o = n0 + wn * 32 + nt * 16 + (lane & 15);
        outb[(size_t)s * HID + o] = acc[mt][nt][r] + bias[o];
      }
}

// ---------------------------------------------------------------------------
extern "C" void kernel_launch(void* const* d_in, const int* in_sizes, int n_in,
                              void* d_out, int out_size, void* d_ws, size_t ws_size,
                              hipStream_t stream) {
  const float* x    = (const float*)d_in[0];  // (8, 2048, 2048)
  const float* gr   = (const float*)d_in[1];  // (1025, 2048)
  const float* gi   = (const float*)d_in[2];  // (1025, 2048)
  const float* W    = (const float*)d_in[3];  // (2048, 2048)
  const float* bias = (const float*)d_in[4];  // (2048,)
  float* out = (float*)d_out;

  // Workspace layout (f16 elements), ~36.7 MB total
  _Float16* Cf  = (_Float16*)d_ws;
  _Float16* Sf  = Cf  + (size_t)FP * SEQ;
  _Float16* Ci  = Sf  + (size_t)FP * SEQ;
  _Float16* Si  = Ci  + (size_t)SEQ * FP;
  _Float16* Yr  = Si  + (size_t)SEQ * FP;
  _Float16* Yi  = Yr  + (size_t)FP * HID;
  _Float16* y16 = Yi  + (size_t)FP * HID;

  dft_setup_kernel<<<(FP * SEQ) / 256, 256, 0, stream>>>(Cf, Sf, Ci, Si);

  for (int b = 0; b < 8; ++b) {
    const float* xb = x + (size_t)b * SEQ * HID;
    float* ob = out + (size_t)b * SEQ * HID;
    fwd_gate_gemm<<<dim3(HID / 64, FP / 128), 256, 0, stream>>>(Cf, Sf, xb, gr, gi, Yr, Yi);
    inv_gemm<<<dim3(HID / 64, SEQ / 128), 256, 0, stream>>>(Ci, Si, Yr, Yi, y16);
    proj_gemm<<<dim3(HID / 64, SEQ / 128), 256, 0, stream>>>(y16, W, bias, ob);
  }
}
